// GraphSAGEModel_55490977464426
// MI455X (gfx1250) — compile-verified, matrix operand order checked by reading, not run on hardware
//
#include <hip/hip_runtime.h>

#define D 64

typedef __attribute__((ext_vector_type(16))) _Float16 v16h;
typedef __attribute__((ext_vector_type(8)))  float    v8f;
typedef __attribute__((ext_vector_type(4)))  float    v4f;
typedef __attribute__((ext_vector_type(4)))  unsigned u4v;

union Frag16 { v16h h; u4v q[2]; };

// ---------------------------------------------------------------------------
// Utility: zero a float buffer
// ---------------------------------------------------------------------------
__global__ void fill_zero_kernel(float* __restrict__ p, long long n) {
    long long i = (long long)blockIdx.x * blockDim.x + threadIdx.x;
    if (i < n) p[i] = 0.0f;
}

// ---------------------------------------------------------------------------
// Degree (bidirectional): deg[e1[j]] += 1, deg[e0[j]] += 1
// ---------------------------------------------------------------------------
__global__ void degree_kernel(const long long* __restrict__ e0,
                              const long long* __restrict__ e1,
                              float* __restrict__ deg, int E) {
    int j = blockIdx.x * blockDim.x + threadIdx.x;
    if (j >= E) return;
    atomicAdd(&deg[(int)e1[j]], 1.0f);
    atomicAdd(&deg[(int)e0[j]], 1.0f);
}

// ---------------------------------------------------------------------------
// Scatter-add: for each directed edge, agg[dst] += x[src]  (64 floats)
// One lane per (edge-direction, 16B quarter): b128 gather + 4 no-return
// f32 atomics. Whole working set is L2-resident (~130MB < 192MB L2).
// ---------------------------------------------------------------------------
__global__ void scatter_kernel(const float* __restrict__ x,
                               const long long* __restrict__ e0,
                               const long long* __restrict__ e1,
                               float* __restrict__ agg, int E) {
    long long t = (long long)blockIdx.x * blockDim.x + threadIdx.x;
    long long total = (long long)E * 32;   // 2 directions * 16 quads
    if (t >= total) return;
    int q = (int)(t & 15);
    long long ed = t >> 4;                 // 0 .. 2E-1
    int s, d;
    if (ed < E) {
        int j = (int)ed;
        __builtin_prefetch(&e0[j + 4096], 0, 0);   // global_prefetch_b8
        s = (int)e0[j]; d = (int)e1[j];
    } else {
        int j = (int)(ed - E);
        s = (int)e1[j]; d = (int)e0[j];
    }
    const float4 v = ((const float4*)(x + (long long)s * D))[q];
    float* ap = agg + (long long)d * D + q * 4;
    atomicAdd(ap + 0, v.x);
    atomicAdd(ap + 1, v.y);
    atomicAdd(ap + 2, v.z);
    atomicAdd(ap + 3, v.w);
}

// ---------------------------------------------------------------------------
// Fragment gather (wave32 f16 WMMA layouts, ISA 7.12.2).
// Per lane the 8 fragment dwords are two contiguous 4-dword runs in LDS:
// half offsets kb+{0,2,4,6} and kb+{16,18,20,22}  ->  two ds_load_b128.
// ---------------------------------------------------------------------------
__device__ __forceinline__ void load_frag_pair(const _Float16* lds, int rowbase,
                                               int kb, Frag16& f0, Frag16& f1) {
    const u4v* p = (const u4v*)lds;            // 8 halfs per uint4
    int b = (rowbase + kb) >> 3;               // uint4 index, 16B aligned
    f0.q[0] = p[b];
    f0.q[1] = p[b + 2];
    f1.q[0] = p[b + 4];                        // K0 = 32  (+32 halfs = +4 uint4)
    f1.q[1] = p[b + 6];
}

// ---------------------------------------------------------------------------
// Async bulk copy: nbytes16 chunks of 16B from global gsrc into LDS at ldst.
// Tracked by ASYNCcnt; caller must s_wait_asynccnt + barrier before reading.
// ---------------------------------------------------------------------------
__device__ __forceinline__ void async_copy_b128(const float* gsrc, void* ldst,
                                                int chunks, int tid, int nthreads) {
    unsigned lbase = (unsigned)(size_t)ldst;
    for (int i = tid; i < chunks; i += nthreads) {
        unsigned loff = lbase + (unsigned)i * 16u;
        const float* gp = gsrc + (size_t)i * 4;
        asm volatile("global_load_async_to_lds_b128 %0, %1, off"
                     :: "v"(loff), "v"(gp)
                     : "memory");
    }
}
__device__ __forceinline__ void wait_async0() {
    asm volatile("s_wait_asynccnt 0x0" ::: "memory");
}

// ---------------------------------------------------------------------------
// SAGE layer GEMM: out = relu(x @ Ws + (agg/max(deg,1)) @ Wn + b)
// 64-row tile per block, 4 wave32s, each wave = 16-row strip, 4 col-tiles.
// Weights arrive via async LDS copy overlapped with activation staging.
// ---------------------------------------------------------------------------
__global__ void __launch_bounds__(128)
sage_gemm_kernel(const float* __restrict__ xin, const float* __restrict__ agg,
                 const float* __restrict__ deg, const float* __restrict__ Ws,
                 const float* __restrict__ Wn, const float* __restrict__ bias,
                 float* __restrict__ xout, int N) {
    __shared__ float    wraw[2 * D * D];   // 32KB raw f32 Ws|Wn (async landing)
    __shared__ _Float16 xs [D * D];        // x tile, f16
    __shared__ _Float16 asx[D * D];        // agg/deg tile, f16
    __shared__ _Float16 wst[D * D];        // Ws^T f16 (n-major)
    __shared__ _Float16 wnt[D * D];        // Wn^T f16
    __shared__ float    outs[D * D];       // f32 result tile
    __shared__ float    bsh[D];

    const int tid = threadIdx.x;
    const int rowBase = blockIdx.x * 64;
    const bool fullTile = (rowBase + 64) <= N;

    // 1) kick off async weight copy (both matrices, 32KB)
    async_copy_b128(Ws, wraw,         1024, tid, 128);
    async_copy_b128(Wn, wraw + D * D, 1024, tid, 128);

    // 2) overlap: bias + activation tiles (f32 -> f16, deg-normalized)
    if (tid < D) bsh[tid] = bias[tid];
    for (int i = tid; i < D * D; i += 128) {
        int r = i >> 6, k = i & 63;
        int row = rowBase + r;
        float xv = 0.0f, av = 0.0f;
        if (fullTile || row < N) {
            xv = xin[(long long)row * D + k];
            float dg = deg[row];
            av = agg[(long long)row * D + k] / (dg > 1.0f ? dg : 1.0f);
        }
        xs[i]  = (_Float16)xv;
        asx[i] = (_Float16)av;
    }

    // 3) weights landed? -> transpose + convert to f16
    wait_async0();
    __syncthreads();
    for (int i = tid; i < D * D; i += 128) {
        int k = i >> 6, n = i & 63;
        wst[n * D + k] = (_Float16)wraw[i];
        wnt[n * D + k] = (_Float16)wraw[D * D + i];
    }
    __syncthreads();

    const int lane = tid & 31;
    const int wave = tid >> 5;
    const int m0   = wave * 16;
    const int ml   = lane & 15;
    const int kb   = (lane >> 4) << 3;

    Frag16 ax0, ax1, aa0, aa1;
    load_frag_pair(xs,  (m0 + ml) * D, kb, ax0, ax1);
    load_frag_pair(asx, (m0 + ml) * D, kb, aa0, aa1);

#pragma unroll
    for (int ct = 0; ct < 4; ++ct) {
        Frag16 bs0, bs1, bn0, bn1;
        int nbase = (ct * 16 + ml) * D;
        load_frag_pair(wst, nbase, kb, bs0, bs1);
        load_frag_pair(wnt, nbase, kb, bn0, bn1);

        v8f acc = {};
        acc = __builtin_amdgcn_wmma_f32_16x16x32_f16(false, ax0.h, false, bs0.h, (short)0, acc, false, false);
        acc = __builtin_amdgcn_wmma_f32_16x16x32_f16(false, ax1.h, false, bs1.h, (short)0, acc, false, false);
        acc = __builtin_amdgcn_wmma_f32_16x16x32_f16(false, aa0.h, false, bn0.h, (short)0, acc, false, false);
        acc = __builtin_amdgcn_wmma_f32_16x16x32_f16(false, aa1.h, false, bn1.h, (short)0, acc, false, false);

        int col = ct * 16 + ml;
        int rb  = m0 + ((lane >> 4) << 3);
#pragma unroll
        for (int r = 0; r < 8; ++r)
            outs[(rb + r) * D + col] = acc[r];
    }
    __syncthreads();

    // 4) coalesced b128 copy-out with fused bias + relu
    for (int i = tid; i < 1024; i += 128) {
        int row = i >> 4;          // 0..63
        int q   = i & 15;          // float4 column quad
        int grow = rowBase + row;
        if (fullTile || grow < N) {
            v4f v  = ((const v4f*)outs)[i];
            v4f bq = ((const v4f*)bsh)[q];
            v = v + bq;
            v.x = v.x > 0.0f ? v.x : 0.0f;
            v.y = v.y > 0.0f ? v.y : 0.0f;
            v.z = v.z > 0.0f ? v.z : 0.0f;
            v.w = v.w > 0.0f ? v.w : 0.0f;
            ((v4f*)(xout + (long long)grow * D))[q] = v;
        }
    }
}

// ---------------------------------------------------------------------------
// Head: out[row] = relu(x @ Wh1 + bh1) @ Wh2 + bh2, fully fused.
// Per-lane dot with Wh2 in the WMMA epilogue, shfl_xor reduce over N lanes.
// ---------------------------------------------------------------------------
__global__ void __launch_bounds__(128)
head_kernel(const float* __restrict__ x, const float* __restrict__ Wh1,
            const float* __restrict__ bh1, const float* __restrict__ Wh2,
            const float* __restrict__ bh2, float* __restrict__ out, int N) {
    __shared__ float    wraw[D * D];   // 16KB raw f32 Wh1 (async landing)
    __shared__ _Float16 xs[D * D];
    __shared__ _Float16 wt[D * D];

    const int tid = threadIdx.x;
    const int rowBase = blockIdx.x * 64;
    const bool fullTile = (rowBase + 64) <= N;

    async_copy_b128(Wh1, wraw, 1024, tid, 128);

    for (int i = tid; i < D * D; i += 128) {
        int r = i >> 6, k = i & 63;
        int row = rowBase + r;
        xs[i] = (_Float16)((fullTile || row < N) ? x[(long long)row * D + k] : 0.0f);
    }

    wait_async0();
    __syncthreads();
    for (int i = tid; i < D * D; i += 128) {
        int k = i >> 6, n = i & 63;
        wt[n * D + k] = (_Float16)wraw[i];
    }
    __syncthreads();

    const int lane = tid & 31;
    const int wave = tid >> 5;
    const int m0   = wave * 16;
    const int ml   = lane & 15;
    const int kb   = (lane >> 4) << 3;

    Frag16 a0, a1;
    load_frag_pair(xs, (m0 + ml) * D, kb, a0, a1);

    float partial[8] = {0.f, 0.f, 0.f, 0.f, 0.f, 0.f, 0.f, 0.f};

#pragma unroll
    for (int ct = 0; ct < 4; ++ct) {
        Frag16 b0, b1;
        load_frag_pair(wt, (ct * 16 + ml) * D, kb, b0, b1);

        v8f acc = {};
        acc = __builtin_amdgcn_wmma_f32_16x16x32_f16(false, a0.h, false, b0.h, (short)0, acc, false, false);
        acc = __builtin_amdgcn_wmma_f32_16x16x32_f16(false, a1.h, false, b1.h, (short)0, acc, false, false);

        int col = ct * 16 + ml;
        float b1v = bh1[col];
        float w2v = Wh2[col];
#pragma unroll
        for (int r = 0; r < 8; ++r) {
            float v = acc[r] + b1v;
            v = v > 0.0f ? v : 0.0f;
            partial[r] += v * w2v;
        }
    }

    // reduce across the 16 N-lanes (masks < 16 stay within each lane-half)
#pragma unroll
    for (int mask = 8; mask > 0; mask >>= 1) {
#pragma unroll
        for (int r = 0; r < 8; ++r)
            partial[r] += __shfl_xor(partial[r], mask, 32);
    }

    if (ml == 0) {
        float bb = bh2[0];
#pragma unroll
        for (int r = 0; r < 8; ++r) {
            int row = rowBase + m0 + ((lane >> 4) << 3) + r;
            if (row < N) out[row] = partial[r] + bb;
        }
    }
}

// ---------------------------------------------------------------------------
// Launch
// ---------------------------------------------------------------------------
extern "C" void kernel_launch(void* const* d_in, const int* in_sizes, int n_in,
                              void* d_out, int out_size, void* d_ws, size_t ws_size,
                              hipStream_t stream) {
    const float*     node = (const float*)d_in[0];
    const long long* eidx = (const long long*)d_in[1];
    const float* Ws0 = (const float*)d_in[2];
    const float* Wn0 = (const float*)d_in[3];
    const float* b0  = (const float*)d_in[4];
    const float* Ws1 = (const float*)d_in[5];
    const float* Wn1 = (const float*)d_in[6];
    const float* b1  = (const float*)d_in[7];
    const float* Wh1 = (const float*)d_in[8];
    const float* bh1 = (const float*)d_in[9];
    const float* Wh2 = (const float*)d_in[10];
    const float* bh2 = (const float*)d_in[11];

    const int N = in_sizes[0] / D;
    const int E = in_sizes[1] / 2;
    const long long* e0 = eidx;
    const long long* e1 = eidx + E;

    float* x1  = (float*)d_ws;
    float* x2  = x1  + (size_t)N * D;
    float* agg = x2  + (size_t)N * D;
    float* deg = agg + (size_t)N * D;   // agg & deg contiguous

    const long long aggdeg = (long long)N * D + N;
    const long long aggn   = (long long)N * D;
    const long long sc_total = (long long)E * 32;
    const int gblocks = (N + 63) / 64;

    // zero agg + deg in one pass (contiguous)
    fill_zero_kernel<<<(int)((aggdeg + 255) / 256), 256, 0, stream>>>(agg, aggdeg);
    degree_kernel<<<(E + 255) / 256, 256, 0, stream>>>(e0, e1, deg, E);

    // layer 0
    scatter_kernel<<<(int)((sc_total + 255) / 256), 256, 0, stream>>>(node, e0, e1, agg, E);
    sage_gemm_kernel<<<gblocks, 128, 0, stream>>>(node, agg, deg, Ws0, Wn0, b0, x1, N);

    // layer 1
    fill_zero_kernel<<<(int)((aggn + 255) / 256), 256, 0, stream>>>(agg, aggn);
    scatter_kernel<<<(int)((sc_total + 255) / 256), 256, 0, stream>>>(x1, e0, e1, agg, E);
    sage_gemm_kernel<<<gblocks, 128, 0, stream>>>(x1, agg, deg, Ws1, Wn1, b1, x2, N);

    // fused head
    head_kernel<<<gblocks, 128, 0, stream>>>(x2, Wh1, bh1, Wh2, bh2, (float*)d_out, N);
}
